// MLPAttnGNN_317827580824
// MI455X (gfx1250) — matrix operand order repaired and checked by reading, not compiled, not run on hardware
//
#include <hip/hip_runtime.h>
#include <hip/hip_bf16.h>
#include <math.h>

#define Nn 20000
#define Ee 320000
#define NDIM 128
#define PDIM 64
#define HDIM 128
#define NHEAD 4
#define EPSf 1e-5f

typedef __bf16 bf16;
typedef __attribute__((ext_vector_type(16))) __bf16 v16bf;
typedef __attribute__((ext_vector_type(8)))  float  v8f;

// ---------- helpers ----------

__device__ __forceinline__ bf16 f2bf(float f) {
  unsigned u = __float_as_uint(f);
  unsigned r = (u + 0x7FFFu + ((u >> 16) & 1u)) >> 16;   // round-nearest-even
  return __builtin_bit_cast(bf16, (unsigned short)r);
}

__device__ __forceinline__ float gelu_f(float x) {
  return 0.5f * x * (1.0f + erff(x * 0.70710678118654752f));
}

// ordered-int trick for float atomic max (nmax initialized to -inf)
__device__ __forceinline__ void atomicMaxFloat(float* addr, float val) {
  int iv = __float_as_int(val);
  if (iv >= 0) atomicMax((int*)addr, iv);
  else         atomicMin((unsigned int*)addr, (unsigned int)iv);
}

// A-fragment swizzle: element (m,k) of a 16x32 bf16 A tile -> flat LDS index.
// Layout per ISA 7.12.2: lane = half*16+m ; VGPR r holds K pairs
//   r<4: K = half*8 + 2r + p ; r>=4: K = 16 + half*8 + 2(r-4) + p
__device__ __forceinline__ int a_swz(int m, int k) {
  int kt = k >> 5, kk = k & 31;
  int half = (kk >> 3) & 1;
  int r = ((kk >> 4) & 1) * 4 + ((kk & 7) >> 1);
  int e = r * 2 + (kk & 1);
  return ((kt * 32 + half * 16 + m) << 4) + e;
}

// stage row-major f32 weight [din x dout] into fragment-major bf16 LDS:
// flat index = (((kt*(dout/16) + nt)*32 + lane) << 4) + e ; value = W[k,n],
// k = kt*32 + (lane>>4)*8 + (e&7) + (e>>3)*16 ; n = nt*16 + (lane&15)
__device__ __forceinline__ void stage_b(bf16* dst, const float* src, int din, int dout,
                                        int tid, int nthr) {
  int tot = din * dout;
  int ntiles = dout >> 4;
  for (int i = tid; i < tot; i += nthr) {
    int e  = i & 15;
    int ln = (i >> 4) & 31;
    int t  = i >> 9;
    int nt = t % ntiles, kt = t / ntiles;
    int k = kt * 32 + (ln >> 4) * 8 + (e & 7) + (e >> 3) * 16;
    int n = nt * 16 + (ln & 15);
    dst[i] = f2bf(src[k * dout + n]);
  }
}

__device__ __forceinline__ void stage_f(float* dst, const float* src, int n, int tid, int nthr) {
  for (int i = tid; i < n; i += nthr) dst[i] = src[i];
}

__device__ __forceinline__ v16bf frag_a(const bf16* act, int kt, int lane) {
  return *(const v16bf*)(act + ((kt * 32 + lane) << 4));
}
__device__ __forceinline__ v16bf frag_b(const bf16* w, int ntiles, int kt, int nt, int lane) {
  return *(const v16bf*)(w + (((kt * ntiles + nt) * 32 + lane) << 4));
}

template <int DIN>
__device__ __forceinline__ v8f gemm_tile(const bf16* act, const bf16* w, int ntiles,
                                         int nt, int lane) {
  v8f acc = {};
#pragma unroll
  for (int kt = 0; kt < DIN / 32; ++kt) {
    acc = __builtin_amdgcn_wmma_f32_16x16x32_bf16(
        false, frag_a(act, kt, lane), false, frag_b(w, ntiles, kt, nt, lane),
        (short)0, acc, false, false);
  }
  return acc;
}

// epilogue: gelu(acc+bias) written into next layer's A-swizzled bf16 buffer
__device__ __forceinline__ void epi_gelu_to_a(bf16* h, v8f acc, const float* bias,
                                              int nt, int lane) {
  int n = nt * 16 + (lane & 15);
  int half = lane >> 4;
#pragma unroll
  for (int r = 0; r < 8; ++r) {
    int m = r + half * 8;
    h[a_swz(m, n)] = f2bf(gelu_f(acc[r] + bias[n]));
  }
}

// ---------- kernels ----------

__global__ void gnn_init_kernel(float* nmax, float* den, float* ao) {
  int i = blockIdx.x * blockDim.x + threadIdx.x;
  if (i < Nn * NHEAD) { nmax[i] = -INFINITY; den[i] = 0.0f; }
  if (i < Nn * 512)   ao[i] = 0.0f;
}

// ---- K1: edge FFN, z' = z + BN(gelu([ss,sd,z]@e_w1+b1)@e_w2+b2) ----
__global__ void gnn_edge_ffn_kernel(const float* __restrict__ s, const float* __restrict__ z,
                                    const int* __restrict__ ei,
                                    const float* ew1, const float* eb1,
                                    const float* ew2, const float* eb2,
                                    const float* eg, const float* ebb,
                                    const float* em, const float* ev,
                                    float* __restrict__ z_out) {
  extern __shared__ __align__(32) char smem[];
  bf16* w1 = (bf16*)smem;                       // 320*128
  bf16* w2 = w1 + 320 * 128;                    // 128*64
  float* b1 = (float*)(w2 + 128 * 64);          // 128
  float* b2 = b1 + 128;                         // 64
  float* bnp = b2 + 64;                         // 4*64
  bf16* actb = (bf16*)(bnp + 256);              // 8*16*320
  bf16* hb = actb + 8 * 16 * 320;               // 8*16*128

  int tid = threadIdx.x, lane = tid & 31, wv = tid >> 5;
  stage_b(w1, ew1, 320, 128, tid, 256);
  stage_b(w2, ew2, 128, 64, tid, 256);
  stage_f(b1, eb1, 128, tid, 256);
  stage_f(b2, eb2, 64, tid, 256);
  stage_f(bnp, eg, 64, tid, 256); stage_f(bnp + 64, ebb, 64, tid, 256);
  stage_f(bnp + 128, em, 64, tid, 256); stage_f(bnp + 192, ev, 64, tid, 256);

  bf16* act = actb + wv * 16 * 320;
  bf16* h = hb + wv * 16 * 128;
  int base = blockIdx.x * 128 + wv * 16;
  if (lane == 0) __builtin_prefetch(ei + base + 128, 0, 0);

  // stage activations [ss | sd | z] in A-fragment layout
  for (int idx = lane; idx < 16 * 320; idx += 32) {
    int m = idx & 15, c = idx >> 4;
    int e = base + m;
    float v;
    if (c < 128)      v = s[ei[e] * NDIM + c];
    else if (c < 256) v = s[ei[Ee + e] * NDIM + (c - 128)];
    else              v = z[e * PDIM + (c - 256)];
    act[a_swz(m, c)] = f2bf(v);
  }
  __syncthreads();

  for (int nt = 0; nt < 8; ++nt) {
    v8f acc = gemm_tile<320>(act, w1, 8, nt, lane);
    epi_gelu_to_a(h, acc, b1, nt, lane);
  }
  __syncthreads();

  for (int nt = 0; nt < 4; ++nt) {
    v8f acc = gemm_tile<128>(h, w2, 4, nt, lane);
    int n = nt * 16 + (lane & 15), half = lane >> 5 ? 0 : (lane >> 4);
    half = lane >> 4;
    float sc = rsqrtf(bnp[192 + n] + EPSf) * bnp[n];
#pragma unroll
    for (int r = 0; r < 8; ++r) {
      int m = r + half * 8, e = base + m;
      float y = acc[r] + b2[n];
      y = (y - bnp[128 + n]) * sc + bnp[64 + n];
      z_out[e * PDIM + n] = z[e * PDIM + n] + y;
    }
  }
}

// ---- K2: attention logits aw[E,4] + per-dst running max ----
__global__ void gnn_logits_kernel(const float* __restrict__ s, const float* __restrict__ zn,
                                  const int* __restrict__ ei,
                                  const float* ww1, const float* wb1,
                                  const float* ww2, const float* wb2,
                                  const float* ww3, const float* wb3,
                                  float* __restrict__ aw, float* __restrict__ nmax) {
  extern __shared__ __align__(32) char smem[];
  bf16* w1 = (bf16*)smem;                       // 320*128
  bf16* w2 = w1 + 320 * 128;                    // 128*128
  bf16* w3 = w2 + 128 * 128;                    // 128*16 (padded)
  float* b1 = (float*)(w3 + 128 * 16);          // 128
  float* b2 = b1 + 128;                         // 128
  float* b3 = b2 + 128;                         // 16
  bf16* actb = (bf16*)(b3 + 16);                // 8*16*320
  bf16* h1b = actb + 8 * 16 * 320;              // 8*16*128
  bf16* h2b = h1b + 8 * 16 * 128;               // 8*16*128

  int tid = threadIdx.x, lane = tid & 31, wv = tid >> 5;
  stage_b(w1, ww1, 320, 128, tid, 256);
  stage_b(w2, ww2, 128, 128, tid, 256);
  // pad 128x4 -> 128x16 in fragment layout
  for (int i = tid; i < 128 * 16; i += 256) {
    int e = i & 15, ln = (i >> 4) & 31, kt = i >> 9;
    int k = kt * 32 + (ln >> 4) * 8 + (e & 7) + (e >> 3) * 16;
    int n = ln & 15;
    w3[i] = f2bf((n < 4) ? ww3[k * 4 + n] : 0.0f);
  }
  stage_f(b1, wb1, 128, tid, 256);
  stage_f(b2, wb2, 128, tid, 256);
  for (int i = tid; i < 16; i += 256) b3[i] = (i < 4) ? wb3[i] : 0.0f;

  bf16* act = actb + wv * 16 * 320;
  bf16* h1 = h1b + wv * 16 * 128;
  bf16* h2 = h2b + wv * 16 * 128;
  int base = blockIdx.x * 128 + wv * 16;

  // stage [sd | ss | z'] in fragment layout
  for (int idx = lane; idx < 16 * 320; idx += 32) {
    int m = idx & 15, c = idx >> 4;
    int e = base + m;
    float v;
    if (c < 128)      v = s[ei[Ee + e] * NDIM + c];
    else if (c < 256) v = s[ei[e] * NDIM + (c - 128)];
    else              v = zn[e * PDIM + (c - 256)];
    act[a_swz(m, c)] = f2bf(v);
  }
  __syncthreads();

  for (int nt = 0; nt < 8; ++nt)
    epi_gelu_to_a(h1, gemm_tile<320>(act, w1, 8, nt, lane), b1, nt, lane);
  __syncthreads();
  for (int nt = 0; nt < 8; ++nt)
    epi_gelu_to_a(h2, gemm_tile<128>(h1, w2, 8, nt, lane), b2, nt, lane);
  __syncthreads();

  v8f acc = gemm_tile<128>(h2, w3, 1, 0, lane);
  int n = lane & 15, half = lane >> 4;
  if (n < 4) {
#pragma unroll
    for (int r = 0; r < 8; ++r) {
      int m = r + half * 8, e = base + m;
      float v = acc[r] + b3[n];
      aw[e * NHEAD + n] = v;
      atomicMaxFloat(&nmax[ei[Ee + e] * NHEAD + n], v);
    }
  }
}

// ---- K3: value MLP + factored softmax scatter (Σe^w, Σe^w·v) ----
__global__ void gnn_values_kernel(const float* __restrict__ s, const float* __restrict__ zn,
                                  const int* __restrict__ ei,
                                  const float* vw1, const float* vb1,
                                  const float* vw2, const float* vb2,
                                  const float* vw3, const float* vb3,
                                  const float* __restrict__ aw, const float* __restrict__ nmax,
                                  float* __restrict__ den, float* __restrict__ ao) {
  extern __shared__ __align__(32) char smem[];
  bf16* w1 = (bf16*)smem;                       // 192*128
  bf16* w2 = w1 + 192 * 128;                    // 128*128
  bf16* w3 = w2 + 128 * 128;                    // 128*128
  float* b1 = (float*)(w3 + 128 * 128);         // 128
  float* b2 = b1 + 128;                         // 128
  float* b3 = b2 + 128;                         // 128
  bf16* actb = (bf16*)(b3 + 128);               // 8*16*192
  bf16* h1b = actb + 8 * 16 * 192;              // 8*16*128
  bf16* h2b = h1b + 8 * 16 * 128;               // 8*16*128
  float* avb = (float*)(h2b + 8 * 16 * 128);    // 8*16*128 f32
  float* ewb = avb + 8 * 16 * 128;              // 8*16*4  f32

  int tid = threadIdx.x, lane = tid & 31, wv = tid >> 5;
  stage_b(w1, vw1, 192, 128, tid, 256);
  stage_b(w2, vw2, 128, 128, tid, 256);
  stage_b(w3, vw3, 128, 128, tid, 256);
  stage_f(b1, vb1, 128, tid, 256);
  stage_f(b2, vb2, 128, tid, 256);
  stage_f(b3, vb3, 128, tid, 256);

  bf16* act = actb + wv * 16 * 192;
  bf16* h1 = h1b + wv * 16 * 128;
  bf16* h2 = h2b + wv * 16 * 128;
  float* av = avb + wv * 16 * 128;
  float* ew = ewb + wv * 16 * 4;
  int base = blockIdx.x * 128 + wv * 16;

  // stage [ss | z'] in fragment layout
  for (int idx = lane; idx < 16 * 192; idx += 32) {
    int m = idx & 15, c = idx >> 4;
    int e = base + m;
    float v = (c < 128) ? s[ei[e] * NDIM + c] : zn[e * PDIM + (c - 128)];
    act[a_swz(m, c)] = f2bf(v);
  }
  __syncthreads();

  for (int nt = 0; nt < 8; ++nt)
    epi_gelu_to_a(h1, gemm_tile<192>(act, w1, 8, nt, lane), b1, nt, lane);
  __syncthreads();
  for (int nt = 0; nt < 8; ++nt)
    epi_gelu_to_a(h2, gemm_tile<128>(h1, w2, 8, nt, lane), b2, nt, lane);
  __syncthreads();

  for (int nt = 0; nt < 8; ++nt) {
    v8f acc = gemm_tile<128>(h2, w3, 8, nt, lane);
    int n = nt * 16 + (lane & 15), half = lane >> 4;
#pragma unroll
    for (int r = 0; r < 8; ++r) av[(r + half * 8) * 128 + n] = acc[r] + b3[n];
  }

  // exp weights + denominator scatter
  for (int idx = lane; idx < 64; idx += 32) {
    int m = idx >> 2, hh = idx & 3;
    int e = base + m, d = ei[Ee + e];
    float w = expf(aw[e * NHEAD + hh] - nmax[d * NHEAD + hh]);
    ew[m * 4 + hh] = w;
    atomicAdd(&den[d * NHEAD + hh], w);
  }
  __syncthreads();

  // weighted value scatter: ao[d, hh, c] += ew * av
  for (int idx = lane; idx < 16 * 512; idx += 32) {
    int m = idx >> 9, rem = idx & 511;
    int hh = rem >> 7, c = rem & 127;
    int e = base + m, d = ei[Ee + e];
    atomicAdd(&ao[d * 512 + rem], ew[m * 4 + hh] * av[m * 128 + c]);
  }
}

// ---- K4a: s' = s + BN((ao/den) @ o_w + o_b) ----
__global__ void gnn_outproj_kernel(const float* __restrict__ s, const float* __restrict__ ao,
                                   const float* __restrict__ den,
                                   const float* ow, const float* ob,
                                   const float* og, const float* obb,
                                   const float* om, const float* ov,
                                   float* __restrict__ s_out) {
  extern __shared__ __align__(32) char smem[];
  bf16* w = (bf16*)smem;                        // 512*128
  float* b = (float*)(w + 512 * 128);           // 128
  float* bnp = b + 128;                         // 4*128
  bf16* actb = (bf16*)(bnp + 512);              // 8*16*512

  int tid = threadIdx.x, lane = tid & 31, wv = tid >> 5;
  stage_b(w, ow, 512, 128, tid, 256);
  stage_f(b, ob, 128, tid, 256);
  stage_f(bnp, og, 128, tid, 256); stage_f(bnp + 128, obb, 128, tid, 256);
  stage_f(bnp + 256, om, 128, tid, 256); stage_f(bnp + 384, ov, 128, tid, 256);

  bf16* act = actb + wv * 16 * 512;
  int base = blockIdx.x * 128 + wv * 16;

  for (int idx = lane; idx < 16 * 512; idx += 32) {
    int m = idx & 15, c = idx >> 4;
    int node = base + m;
    float v = 0.0f;
    if (node < Nn) {
      float d = den[node * NHEAD + (c >> 7)];
      v = ao[node * 512 + c] / (d > 0.0f ? d : 1.0f);
    }
    act[a_swz(m, c)] = f2bf(v);
  }
  __syncthreads();

  for (int nt = 0; nt < 8; ++nt) {
    v8f acc = gemm_tile<512>(act, w, 8, nt, lane);
    int n = nt * 16 + (lane & 15), half = lane >> 4;
    float sc = rsqrtf(bnp[384 + n] + EPSf) * bnp[n];
#pragma unroll
    for (int r = 0; r < 8; ++r) {
      int m = r + half * 8, node = base + m;
      if (node < Nn) {
        float y = (acc[r] + b[n] - bnp[256 + n]) * sc + bnp[128 + n];
        s_out[node * NDIM + n] = s[node * NDIM + n] + y;
      }
    }
  }
}

// ---- K4b: s'' = s' + BN(gelu(s'@f_w1+b1)@f_w2+b2) ----
__global__ void gnn_nodeffn_kernel(float* __restrict__ s_out,
                                   const float* fw1, const float* fb1,
                                   const float* fw2, const float* fb2,
                                   const float* fg, const float* fbb,
                                   const float* fm, const float* fv) {
  extern __shared__ __align__(32) char smem[];
  bf16* w1 = (bf16*)smem;                       // 128*128
  bf16* w2 = w1 + 128 * 128;                    // 128*128
  float* b1 = (float*)(w2 + 128 * 128);         // 128
  float* b2 = b1 + 128;                         // 128
  float* bnp = b2 + 128;                        // 4*128
  bf16* actb = (bf16*)(bnp + 512);              // 8*16*128
  bf16* hb = actb + 8 * 16 * 128;               // 8*16*128

  int tid = threadIdx.x, lane = tid & 31, wv = tid >> 5;
  stage_b(w1, fw1, 128, 128, tid, 256);
  stage_b(w2, fw2, 128, 128, tid, 256);
  stage_f(b1, fb1, 128, tid, 256);
  stage_f(b2, fb2, 128, tid, 256);
  stage_f(bnp, fg, 128, tid, 256); stage_f(bnp + 128, fbb, 128, tid, 256);
  stage_f(bnp + 256, fm, 128, tid, 256); stage_f(bnp + 384, fv, 128, tid, 256);

  bf16* act = actb + wv * 16 * 128;
  bf16* h = hb + wv * 16 * 128;
  int base = blockIdx.x * 128 + wv * 16;

  for (int idx = lane; idx < 16 * 128; idx += 32) {
    int m = idx & 15, c = idx >> 4;
    int node = base + m;
    float v = (node < Nn) ? s_out[node * NDIM + c] : 0.0f;
    act[a_swz(m, c)] = f2bf(v);
  }
  __syncthreads();

  for (int nt = 0; nt < 8; ++nt)
    epi_gelu_to_a(h, gemm_tile<128>(act, w1, 8, nt, lane), b1, nt, lane);
  __syncthreads();

  for (int nt = 0; nt < 8; ++nt) {
    v8f acc = gemm_tile<128>(h, w2, 8, nt, lane);
    int n = nt * 16 + (lane & 15), half = lane >> 4;
    float sc = rsqrtf(bnp[384 + n] + EPSf) * bnp[n];
#pragma unroll
    for (int r = 0; r < 8; ++r) {
      int m = r + half * 8, node = base + m;
      if (node < Nn) {
        float y = (acc[r] + b2[n] - bnp[256 + n]) * sc + bnp[128 + n];
        s_out[node * NDIM + n] = s_out[node * NDIM + n] + y;
      }
    }
  }
}

// ---------- launch ----------

extern "C" void kernel_launch(void* const* d_in, const int* in_sizes, int n_in,
                              void* d_out, int out_size, void* d_ws, size_t ws_size,
                              hipStream_t stream) {
  (void)in_sizes; (void)n_in; (void)out_size; (void)ws_size;
  const float* s  = (const float*)d_in[0];
  const float* z  = (const float*)d_in[1];
  const int*   ei = (const int*)d_in[2];
  const float* w_w1 = (const float*)d_in[3];  const float* w_b1 = (const float*)d_in[4];
  const float* w_w2 = (const float*)d_in[5];  const float* w_b2 = (const float*)d_in[6];
  const float* w_w3 = (const float*)d_in[7];  const float* w_b3 = (const float*)d_in[8];
  const float* v_w1 = (const float*)d_in[9];  const float* v_b1 = (const float*)d_in[10];
  const float* v_w2 = (const float*)d_in[11]; const float* v_b2 = (const float*)d_in[12];
  const float* v_w3 = (const float*)d_in[13]; const float* v_b3 = (const float*)d_in[14];
  const float* o_w  = (const float*)d_in[15]; const float* o_b  = (const float*)d_in[16];
  const float* o_g  = (const float*)d_in[17]; const float* o_bb = (const float*)d_in[18];
  const float* o_m  = (const float*)d_in[19]; const float* o_v  = (const float*)d_in[20];
  const float* f_w1 = (const float*)d_in[21]; const float* f_b1 = (const float*)d_in[22];
  const float* f_w2 = (const float*)d_in[23]; const float* f_b2 = (const float*)d_in[24];
  const float* f_g  = (const float*)d_in[25]; const float* f_bb = (const float*)d_in[26];
  const float* f_m  = (const float*)d_in[27]; const float* f_v  = (const float*)d_in[28];
  const float* e_w1 = (const float*)d_in[29]; const float* e_b1 = (const float*)d_in[30];
  const float* e_w2 = (const float*)d_in[31]; const float* e_b2 = (const float*)d_in[32];
  const float* e_g  = (const float*)d_in[33]; const float* e_bb = (const float*)d_in[34];
  const float* e_m  = (const float*)d_in[35]; const float* e_v  = (const float*)d_in[36];

  float* s_out = (float*)d_out;
  float* z_out = s_out + (size_t)Nn * NDIM;

  float* aw   = (float*)d_ws;                    // E*4
  float* nmax = aw + (size_t)Ee * NHEAD;         // N*4
  float* den  = nmax + (size_t)Nn * NHEAD;       // N*4
  float* ao   = den + (size_t)Nn * NHEAD;        // N*512

  const int EB = Ee / 128;                       // 2500
  const int NB = (Nn + 127) / 128;               // 157

  size_t sm1 = (320*128 + 128*64) * 2 + (128 + 64 + 256) * 4
             + (8*16*320 + 8*16*128) * 2;
  size_t sm2 = (320*128 + 128*128 + 128*16) * 2 + (128 + 128 + 16) * 4
             + (8*16*320 + 2 * 8*16*128) * 2;
  size_t sm3 = (192*128 + 2*128*128) * 2 + (3*128) * 4
             + (8*16*192 + 2 * 8*16*128) * 2 + (8*16*128 + 8*16*4) * 4;
  size_t sm4 = (512*128) * 2 + (128 + 512) * 4 + (8*16*512) * 2;
  size_t sm5 = (2*128*128) * 2 + (128 + 128 + 512) * 4 + (2 * 8*16*128) * 2;

  gnn_init_kernel<<<(Nn * 512) / 256, 256, 0, stream>>>(nmax, den, ao);
  gnn_edge_ffn_kernel<<<EB, 256, sm1, stream>>>(s, z, ei, e_w1, e_b1, e_w2, e_b2,
                                                e_g, e_bb, e_m, e_v, z_out);
  gnn_logits_kernel<<<EB, 256, sm2, stream>>>(s, z_out, ei, w_w1, w_b1, w_w2, w_b2,
                                              w_w3, w_b3, aw, nmax);
  gnn_values_kernel<<<EB, 256, sm3, stream>>>(s, z_out, ei, v_w1, v_b1, v_w2, v_b2,
                                              v_w3, v_b3, aw, nmax, den, ao);
  gnn_outproj_kernel<<<NB, 256, sm4, stream>>>(s, ao, den, o_w, o_b, o_g, o_bb,
                                               o_m, o_v, s_out);
  gnn_nodeffn_kernel<<<NB, 256, sm5, stream>>>(s_out, f_w1, f_b1, f_w2, f_b2,
                                               f_g, f_bb, f_m, f_v);
}
